// TransformerEncoderLayer_76287209111767
// MI455X (gfx1250) — matrix operand order repaired and checked by reading, not compiled, not run on hardware
//
#include <hip/hip_runtime.h>

typedef _Float16 v16h __attribute__((ext_vector_type(16)));
typedef _Float16 v8h  __attribute__((ext_vector_type(8)));
typedef _Float16 v4h  __attribute__((ext_vector_type(4)));
typedef _Float16 v2h  __attribute__((ext_vector_type(2)));
typedef float    v8f  __attribute__((ext_vector_type(8)));

#define SEQ   1024
#define BATCH 4
#define DIM   1024
#define NHEAD 16
#define HDIM  64
#define DFFN  4096
#define NROWS (SEQ*BATCH)   // 4096

// CDNA5 async global->LDS path (GLOBAL_LOAD_ASYNC_TO_LDS_B128 / ASYNCcnt),
// guarded so the file still builds on toolchains without the builtins.
#if defined(__gfx1250__) && \
    __has_builtin(__builtin_amdgcn_global_load_async_to_lds_b128) && \
    __has_builtin(__builtin_amdgcn_s_wait_asynccnt)
#define USE_ASYNC_LDS 1
#else
#define USE_ASYNC_LDS 0
#endif

#if USE_ASYNC_LDS
// builtin expects pointers to 16B int vectors: as1 (global src), as3 (LDS dst)
typedef int async_v4i __attribute__((vector_size(4 * sizeof(int))));
#define AS1P(p) ((__attribute__((address_space(1))) async_v4i*)(p))
#define AS3P(p) ((__attribute__((address_space(3))) async_v4i*)(p))
#endif

// ---------------------------------------------------------------------------
// WMMA fragment loaders (wave32, 16x16x32 f16).
// A (16x32): lanes 0-15 row M=lane, K pairs {0..7,16..23}; lanes 16-31 same M,
//   K pairs {8..15,24..31}  (CDNA5 ISA 16-bit A table).
// B (32x16): lanes 0-15 col N=lane hold K={0..7,16..23}; lanes 16-31 hold K={8..15,24..31}.
// C/D: vgpr i -> (M = 8*(lane>=16)+i, N = lane&15).
// ---------------------------------------------------------------------------
__device__ __forceinline__ v16h load_fragA(const _Float16* p, int ld) {
    const int lane = threadIdx.x & 31;
    const int row  = lane & 15;
    const int hi   = lane >> 4;
    const _Float16* base = p + row * ld;
    v16h f;
#pragma unroll
    for (int v = 0; v < 8; ++v) {
        int k = (v < 4) ? (hi * 8 + 2 * v) : (16 + hi * 8 + 2 * (v - 4));
        v2h t = *(const v2h*)(base + k);
        f[2 * v]     = t[0];
        f[2 * v + 1] = t[1];
    }
    return f;
}

__device__ __forceinline__ v16h load_fragB(const _Float16* p, int ld) {
    const int lane = threadIdx.x & 31;
    const int n  = lane & 15;
    const int hi = lane >> 4;
    const _Float16* base = p + n * ld + hi * 8;    // 16B aligned by construction
    v8h lo = *(const v8h*)(base);
    v8h hv = *(const v8h*)(base + 16);
    v16h f;
#pragma unroll
    for (int j = 0; j < 8; ++j) { f[j] = lo[j]; f[8 + j] = hv[j]; }
    return f;
}

// ---------------------------------------------------------------------------
// LSQ forward quantization: round(clamp(w/alpha,-8,7))*alpha -> f16 (x4/thread)
// ---------------------------------------------------------------------------
__global__ void quant_kernel(const float* __restrict__ w,
                             const float* __restrict__ alpha,
                             _Float16* __restrict__ out, int n) {
    int i4 = (blockIdx.x * 256 + threadIdx.x) * 4;
    if (i4 + 3 < n) {
        float a  = alpha[0];
        float ra = 1.0f / a;
        float4 wv = *(const float4*)(w + i4);
        v4h o;
        float q;
        q = rintf(fminf(fmaxf(wv.x * ra, -8.0f), 7.0f)); o[0] = (_Float16)(q * a);
        q = rintf(fminf(fmaxf(wv.y * ra, -8.0f), 7.0f)); o[1] = (_Float16)(q * a);
        q = rintf(fminf(fmaxf(wv.z * ra, -8.0f), 7.0f)); o[2] = (_Float16)(q * a);
        q = rintf(fminf(fmaxf(wv.w * ra, -8.0f), 7.0f)); o[3] = (_Float16)(q * a);
        *(v4h*)(out + i4) = o;
    }
}

// ---------------------------------------------------------------------------
// LayerNorm over last dim (1024). o0 = LN(x); o1 (optional) = LN(x)+pos.
// One block (256 thr) per row; 1 float4 per thread.
// ---------------------------------------------------------------------------
template<bool HAS_POS>
__global__ __launch_bounds__(256) void ln_kernel(
    const float* __restrict__ x, const float* __restrict__ pos,
    const float* __restrict__ g, const float* __restrict__ bb,
    _Float16* __restrict__ o0, _Float16* __restrict__ o1) {
    __shared__ float s1[256], s2[256];
    const int r = blockIdx.x, tid = threadIdx.x;
    const int d4 = tid * 4;
    const size_t rowoff = (size_t)r * DIM;
    float4 xv = *(const float4*)(x + rowoff + d4);
    float s  = xv.x + xv.y + xv.z + xv.w;
    float ss = xv.x * xv.x + xv.y * xv.y + xv.z * xv.z + xv.w * xv.w;
    s1[tid] = s; s2[tid] = ss; __syncthreads();
    for (int o = 128; o > 0; o >>= 1) {
        if (tid < o) { s1[tid] += s1[tid + o]; s2[tid] += s2[tid + o]; }
        __syncthreads();
    }
    float mean = s1[0] * (1.0f / DIM);
    float var  = s2[0] * (1.0f / DIM) - mean * mean;
    float rs   = rsqrtf(var + 1e-5f);
    float4 gv = *(const float4*)(g + d4);
    float4 bv = *(const float4*)(bb + d4);
    float y0 = (xv.x - mean) * rs * gv.x + bv.x;
    float y1 = (xv.y - mean) * rs * gv.y + bv.y;
    float y2 = (xv.z - mean) * rs * gv.z + bv.z;
    float y3 = (xv.w - mean) * rs * gv.w + bv.w;
    v4h o;
    o[0] = (_Float16)y0; o[1] = (_Float16)y1;
    o[2] = (_Float16)y2; o[3] = (_Float16)y3;
    *(v4h*)(o0 + rowoff + d4) = o;
    if (HAS_POS) {
        float4 pv = *(const float4*)(pos + rowoff + d4);
        v4h o2;
        o2[0] = (_Float16)(y0 + pv.x); o2[1] = (_Float16)(y1 + pv.y);
        o2[2] = (_Float16)(y2 + pv.z); o2[3] = (_Float16)(y3 + pv.w);
        *(v4h*)(o1 + rowoff + d4) = o2;
    }
}

// ---------------------------------------------------------------------------
// C[M,N] = act( scale * (A[M,K] @ W[N,K]^T) + bias + resid )
// A,W f16 row-major. Block tile 128x128, BK=64, 8 waves; wave tile 32x64.
// Staging: async global->LDS double buffering (CDNA5 path) when available,
// else register double buffering. L2 prefetch runs ahead in both paths.
// ---------------------------------------------------------------------------
template<bool HAS_BIAS, bool HAS_RES, bool RELU, bool OUT_F, bool OUT_H>
__global__ __launch_bounds__(256) void gemm_wmma(
    const _Float16* __restrict__ A, const _Float16* __restrict__ W,
    const float* __restrict__ bias, const float* __restrict__ resid,
    float* __restrict__ Cf, _Float16* __restrict__ Ch,
    int M, int N, int K, float scale) {
    const int tid  = threadIdx.x;
    const int lane = tid & 31;
    const int wid  = tid >> 5;
    const int ln   = lane & 15;
    const int hi   = lane >> 4;
    const int bm = blockIdx.y * 128, bn = blockIdx.x * 128;
    const int m0 = (wid & 3) * 32, n0 = (wid >> 2) * 64;

    // staging addresses: each thread owns 4 chunks of 8 halfs per matrix
    const int c8 = (tid & 7) * 8;        // col within 64-wide slab
    const int r0 = tid >> 3;             // 0..31, rows r0 + 32*it
    const _Float16* pA[4];
    const _Float16* pW[4];
#pragma unroll
    for (int it = 0; it < 4; ++it) {
        pA[it] = A + (size_t)(bm + r0 + it * 32) * K + c8;
        pW[it] = W + (size_t)(bn + r0 + it * 32) * K + c8;
    }

    v8f vzero = {};
    v8f acc[2][4];
#pragma unroll
    for (int tm = 0; tm < 2; ++tm)
#pragma unroll
        for (int tn = 0; tn < 4; ++tn) acc[tm][tn] = vzero;

    const int nk = K >> 6;

#if USE_ASYNC_LDS
    __shared__ alignas(16) _Float16 As[2][128][72];
    __shared__ alignas(16) _Float16 Ws[2][128][72];
    // issue stage 0
#pragma unroll
    for (int it = 0; it < 4; ++it) {
        __builtin_amdgcn_global_load_async_to_lds_b128(
            AS1P(pA[it]), AS3P(&As[0][r0 + it * 32][c8]), 0, 0);
        __builtin_amdgcn_global_load_async_to_lds_b128(
            AS1P(pW[it]), AS3P(&Ws[0][r0 + it * 32][c8]), 0, 0);
        pA[it] += 64; pW[it] += 64;
    }
    int buf = 0;
    for (int kt = 0; kt < nk; ++kt) {
        __builtin_amdgcn_s_wait_asynccnt(0);   // this wave's stage-kt data in LDS
        __syncthreads();                       // everyone's stage-kt data in LDS
        if (kt + 1 < nk) {                     // stream next slab during WMMAs
#pragma unroll
            for (int it = 0; it < 4; ++it) {
                __builtin_amdgcn_global_load_async_to_lds_b128(
                    AS1P(pA[it]), AS3P(&As[buf ^ 1][r0 + it * 32][c8]), 0, 0);
                __builtin_amdgcn_global_load_async_to_lds_b128(
                    AS1P(pW[it]), AS3P(&Ws[buf ^ 1][r0 + it * 32][c8]), 0, 0);
                pA[it] += 64; pW[it] += 64;
            }
            if (kt + 2 < nk) {
                __builtin_prefetch(pA[0], 0, 1);
                __builtin_prefetch(pW[0], 0, 1);
            }
        }
#pragma unroll
        for (int ks = 0; ks < 2; ++ks) {
            v16h af0 = load_fragA(&As[buf][m0][ks * 32], 72);
            v16h af1 = load_fragA(&As[buf][m0 + 16][ks * 32], 72);
#pragma unroll
            for (int tn = 0; tn < 4; ++tn) {
                v16h bf = load_fragB(&Ws[buf][n0 + tn * 16][ks * 32], 72);
                acc[0][tn] = __builtin_amdgcn_wmma_f32_16x16x32_f16(
                    false, af0, false, bf, (short)0, acc[0][tn], false, false);
                acc[1][tn] = __builtin_amdgcn_wmma_f32_16x16x32_f16(
                    false, af1, false, bf, (short)0, acc[1][tn], false, false);
            }
        }
        buf ^= 1;
    }
#else
    __shared__ alignas(16) _Float16 As[128][72];
    __shared__ alignas(16) _Float16 Ws[128][72];
    uint4 ra[4], rw[4];
#pragma unroll
    for (int it = 0; it < 4; ++it) {
        ra[it] = *(const uint4*)pA[it]; pA[it] += 64;
        rw[it] = *(const uint4*)pW[it]; pW[it] += 64;
    }
    for (int kt = 0; kt < nk; ++kt) {
        __syncthreads();
#pragma unroll
        for (int it = 0; it < 4; ++it) {
            *(uint4*)&As[r0 + it * 32][c8] = ra[it];
            *(uint4*)&Ws[r0 + it * 32][c8] = rw[it];
        }
        __syncthreads();
        if (kt + 1 < nk) {       // stage next slab while WMMAs run
#pragma unroll
            for (int it = 0; it < 4; ++it) {
                ra[it] = *(const uint4*)pA[it]; pA[it] += 64;
                rw[it] = *(const uint4*)pW[it]; pW[it] += 64;
            }
            if (kt + 2 < nk) {
                __builtin_prefetch(pA[0] + 64, 0, 1);
                __builtin_prefetch(pW[0] + 64, 0, 1);
            }
        }
#pragma unroll
        for (int ks = 0; ks < 2; ++ks) {
            v16h af0 = load_fragA(&As[m0][ks * 32], 72);
            v16h af1 = load_fragA(&As[m0 + 16][ks * 32], 72);
#pragma unroll
            for (int tn = 0; tn < 4; ++tn) {
                v16h bf = load_fragB(&Ws[n0 + tn * 16][ks * 32], 72);
                acc[0][tn] = __builtin_amdgcn_wmma_f32_16x16x32_f16(
                    false, af0, false, bf, (short)0, acc[0][tn], false, false);
                acc[1][tn] = __builtin_amdgcn_wmma_f32_16x16x32_f16(
                    false, af1, false, bf, (short)0, acc[1][tn], false, false);
            }
        }
    }
#endif

    // branch-free epilogue with incremental addressing
#pragma unroll
    for (int tm = 0; tm < 2; ++tm)
#pragma unroll
        for (int tn = 0; tn < 4; ++tn) {
            const int mbase = bm + m0 + tm * 16 + hi * 8;
            const int nidx  = bn + n0 + tn * 16 + ln;
            const float bv  = HAS_BIAS ? bias[nidx] : 0.0f;
            size_t idx = (size_t)mbase * N + nidx;
#pragma unroll
            for (int i = 0; i < 8; ++i) {
                float v = acc[tm][tn][i] * scale + bv;
                if (HAS_RES) v += resid[idx];
                if (RELU)    v = fmaxf(v, 0.0f);
                if (OUT_F)   Cf[idx] = v;
                if (OUT_H)   Ch[idx] = (_Float16)v;
                idx += N;
            }
        }
}

// ---------------------------------------------------------------------------
// Flash attention: grid (S/64, B*H), 128 threads (4 waves).
// Q,K,V,O f16 in [s][b][h*64+d] layout. Q pre-scaled by 1/sqrt(HD).
// Each wave owns 16 query rows; online softmax over 64-col K/V tiles.
// ---------------------------------------------------------------------------
__global__ __launch_bounds__(128) void attn_kernel(
    const _Float16* __restrict__ Q, const _Float16* __restrict__ Km,
    const _Float16* __restrict__ V, _Float16* __restrict__ O) {
    __shared__ alignas(16) _Float16 Qs[64][72];
    __shared__ alignas(16) _Float16 Ks[64][72];
    __shared__ alignas(16) _Float16 Vs[64][72];   // transposed: [d][kj]
    __shared__ alignas(16) _Float16 Ps[64][72];
    const int tid  = threadIdx.x;
    const int lane = tid & 31;
    const int w    = tid >> 5;
    const int ln   = lane & 15;
    const int hi   = lane >> 4;
    const int qb = blockIdx.x * 64;
    const int bh = blockIdx.y;
    const int b  = bh >> 4;
    const int h  = bh & 15;
    const size_t headoff = (size_t)h * HDIM;

    const int lrow = tid >> 3;          // 0..15, rows lrow + 16*it
    const int lc8  = (tid & 7) * 8;
#pragma unroll
    for (int it = 0; it < 4; ++it) {
        int row = lrow + it * 16;
        *(uint4*)&Qs[row][lc8] = *(const uint4*)(
            Q + ((size_t)(qb + row) * BATCH + b) * DIM + headoff + lc8);
    }
    __syncthreads();
    v16h aq[2];
    aq[0] = load_fragA(&Qs[w * 16][0], 72);
    aq[1] = load_fragA(&Qs[w * 16][32], 72);

    v8f vzero = {};
    v8f acc_o[4];
#pragma unroll
    for (int tn = 0; tn < 4; ++tn) acc_o[tn] = vzero;
    float mst[8], lst[8];
#pragma unroll
    for (int i = 0; i < 8; ++i) { mst[i] = -3.0e38f; lst[i] = 0.0f; }

    for (int kt = 0; kt < SEQ / 64; ++kt) {
        __syncthreads();
#pragma unroll
        for (int it = 0; it < 4; ++it) {
            int row = lrow + it * 16;
            size_t base = ((size_t)(kt * 64 + row) * BATCH + b) * DIM + headoff + lc8;
            *(uint4*)&Ks[row][lc8] = *(const uint4*)(Km + base);
            union { uint4 u; _Float16 hv[8]; } vv;
            vv.u = *(const uint4*)(V + base);
#pragma unroll
            for (int j = 0; j < 8; ++j) Vs[lc8 + j][row] = vv.hv[j];
        }
        __syncthreads();

        v8f sc[4];
#pragma unroll
        for (int tn = 0; tn < 4; ++tn) sc[tn] = vzero;
#pragma unroll
        for (int ks = 0; ks < 2; ++ks)
#pragma unroll
            for (int tn = 0; tn < 4; ++tn) {
                v16h bk = load_fragB(&Ks[tn * 16][ks * 32], 72);
                sc[tn] = __builtin_amdgcn_wmma_f32_16x16x32_f16(
                    false, aq[ks], false, bk, (short)0, sc[tn], false, false);
            }

        // online softmax: row m = hi*8+i lives across the 16 lanes of a half
#pragma unroll
        for (int i = 0; i < 8; ++i) {
            float mx = fmaxf(fmaxf(sc[0][i], sc[1][i]), fmaxf(sc[2][i], sc[3][i]));
#pragma unroll
            for (int d = 1; d < 16; d <<= 1) mx = fmaxf(mx, __shfl_xor(mx, d, 32));
            float nm   = fmaxf(mst[i], mx);
            float corr = __expf(mst[i] - nm);
            float ls = 0.0f;
#pragma unroll
            for (int tn = 0; tn < 4; ++tn) {
                float p = __expf(sc[tn][i] - nm);
                sc[tn][i] = p;
                ls += p;
            }
#pragma unroll
            for (int d = 1; d < 16; d <<= 1) ls += __shfl_xor(ls, d, 32);
            lst[i] = lst[i] * corr + ls;
            mst[i] = nm;
#pragma unroll
            for (int tn = 0; tn < 4; ++tn) acc_o[tn][i] = acc_o[tn][i] * corr;
            int rowP = w * 16 + hi * 8 + i;
#pragma unroll
            for (int tn = 0; tn < 4; ++tn)
                Ps[rowP][tn * 16 + ln] = (_Float16)sc[tn][i];
        }

        // O += P @ V   (Vs holds V^T so fragB reads B[k=kj][n=d])
#pragma unroll
        for (int ks = 0; ks < 2; ++ks) {
            v16h ap = load_fragA(&Ps[w * 16][ks * 32], 72);
#pragma unroll
            for (int tn = 0; tn < 4; ++tn) {
                v16h bv = load_fragB(&Vs[tn * 16][ks * 32], 72);
                acc_o[tn] = __builtin_amdgcn_wmma_f32_16x16x32_f16(
                    false, ap, false, bv, (short)0, acc_o[tn], false, false);
            }
        }
    }

    // writeout: one reciprocal per row, incremental addressing (step BATCH*DIM)
    float rinv[8];
#pragma unroll
    for (int i = 0; i < 8; ++i) rinv[i] = 1.0f / lst[i];
#pragma unroll
    for (int tn = 0; tn < 4; ++tn) {
        size_t idx = ((size_t)(qb + w * 16 + hi * 8) * BATCH + b) * DIM
                   + headoff + tn * 16 + ln;
#pragma unroll
        for (int i = 0; i < 8; ++i) {
            O[idx] = (_Float16)(acc_o[tn][i] * rinv[i]);
            idx += (size_t)BATCH * DIM;
        }
    }
}

// ---------------------------------------------------------------------------
extern "C" void kernel_launch(void* const* d_in, const int* in_sizes, int n_in,
                              void* d_out, int out_size, void* d_ws, size_t ws_size,
                              hipStream_t stream) {
    (void)in_sizes; (void)n_in; (void)out_size; (void)ws_size;
    const float* src        = (const float*)d_in[0];
    const float* pos        = (const float*)d_in[1];
    const float* in_proj_w  = (const float*)d_in[2];
    const float* alpha_in   = (const float*)d_in[3];
    const float* out_proj_w = (const float*)d_in[4];
    const float* alpha_out  = (const float*)d_in[5];
    const float* w1         = (const float*)d_in[6];
    const float* b1         = (const float*)d_in[7];
    const float* alpha1     = (const float*)d_in[8];
    const float* w2         = (const float*)d_in[9];
    const float* b2         = (const float*)d_in[10];
    const float* alpha2     = (const float*)d_in[11];
    const float* ln1_g      = (const float*)d_in[12];
    const float* ln1_b      = (const float*)d_in[13];
    const float* ln2_g      = (const float*)d_in[14];
    const float* ln2_b      = (const float*)d_in[15];

    char* ws = (char*)d_ws;
    size_t off = 0;
    auto alloc = [&](size_t bytes) -> void* {
        void* p = ws + off;
        off += (bytes + 255) & ~(size_t)255;
        return p;
    };
    _Float16* wqkv_h   = (_Float16*)alloc(2ull * 3 * DIM * DIM);
    _Float16* wo_h     = (_Float16*)alloc(2ull * DIM * DIM);
    _Float16* w1_h     = (_Float16*)alloc(2ull * DFFN * DIM);
    _Float16* w2_h     = (_Float16*)alloc(2ull * DIM * DFFN);
    _Float16* qk_h     = (_Float16*)alloc(2ull * NROWS * DIM);
    _Float16* s2_h     = (_Float16*)alloc(2ull * NROWS * DIM);
    _Float16* q_h      = (_Float16*)alloc(2ull * NROWS * DIM);
    _Float16* k_h      = (_Float16*)alloc(2ull * NROWS * DIM);
    _Float16* v_h      = (_Float16*)alloc(2ull * NROWS * DIM);
    _Float16* ctx_h    = (_Float16*)alloc(2ull * NROWS * DIM);
    float*    src_attn = (float*)   alloc(4ull * NROWS * DIM);
    _Float16* s2b_h    = (_Float16*)alloc(2ull * NROWS * DIM);
    _Float16* h_h      = (_Float16*)alloc(2ull * NROWS * DFFN);

    // 1) quantize weights -> f16 (4 elems/thread)
    quant_kernel<<<(3 * DIM * DIM) / 1024, 256, 0, stream>>>(in_proj_w, alpha_in, wqkv_h, 3 * DIM * DIM);
    quant_kernel<<<(DIM * DIM) / 1024, 256, 0, stream>>>(out_proj_w, alpha_out, wo_h, DIM * DIM);
    quant_kernel<<<(DFFN * DIM) / 1024, 256, 0, stream>>>(w1, alpha1, w1_h, DFFN * DIM);
    quant_kernel<<<(DIM * DFFN) / 1024, 256, 0, stream>>>(w2, alpha2, w2_h, DIM * DFFN);

    // 2) LN1 -> src2 (f16) and qk = src2 + pos (f16)
    ln_kernel<true><<<NROWS, 256, 0, stream>>>(src, pos, ln1_g, ln1_b, s2_h, qk_h);

    // 3) QKV projections (q pre-scaled by 1/sqrt(HD)=0.125)
    dim3 gD(DIM / 128, NROWS / 128);
    gemm_wmma<false,false,false,false,true><<<gD, 256, 0, stream>>>(
        qk_h, wqkv_h,                 nullptr, nullptr, nullptr, q_h, NROWS, DIM, DIM, 0.125f);
    gemm_wmma<false,false,false,false,true><<<gD, 256, 0, stream>>>(
        qk_h, wqkv_h + DIM * DIM,     nullptr, nullptr, nullptr, k_h, NROWS, DIM, DIM, 1.0f);
    gemm_wmma<false,false,false,false,true><<<gD, 256, 0, stream>>>(
        s2_h, wqkv_h + 2 * DIM * DIM, nullptr, nullptr, nullptr, v_h, NROWS, DIM, DIM, 1.0f);

    // 4) attention
    attn_kernel<<<dim3(SEQ / 64, BATCH * NHEAD), 128, 0, stream>>>(q_h, k_h, v_h, ctx_h);

    // 5) out-proj + residual -> src_attn (f32)
    gemm_wmma<false,true,false,true,false><<<gD, 256, 0, stream>>>(
        ctx_h, wo_h, nullptr, src, src_attn, nullptr, NROWS, DIM, DIM, 1.0f);

    // 6) LN2 -> s2b (f16)
    ln_kernel<false><<<NROWS, 256, 0, stream>>>(src_attn, nullptr, ln2_g, ln2_b, s2b_h, nullptr);

    // 7) FFN1 with bias + relu -> h (f16)
    dim3 gF(DFFN / 128, NROWS / 128);
    gemm_wmma<true,false,true,false,true><<<gF, 256, 0, stream>>>(
        s2b_h, w1_h, b1, nullptr, nullptr, h_h, NROWS, DFFN, DIM, 1.0f);

    // 8) FFN2 with bias + residual -> out (f32)
    gemm_wmma<true,true,false,true,false><<<gD, 256, 0, stream>>>(
        h_h, w2_h, b2, src_attn, (float*)d_out, nullptr, NROWS, DIM, DFFN, 1.0f);
}
